// GATLayer_41214506172668
// MI455X (gfx1250) — matrix operand order repaired and checked by reading, not compiled, not run on hardware
//
#include <hip/hip_runtime.h>

typedef __attribute__((ext_vector_type(16))) _Float16 v16h;
typedef __attribute__((ext_vector_type(8)))  _Float16 v8h;
typedef __attribute__((ext_vector_type(8)))  float    v8f;
typedef __attribute__((ext_vector_type(4)))  unsigned int v4u;
typedef __attribute__((ext_vector_type(8)))  int      v8i;
typedef __attribute__((ext_vector_type(4)))  int      v4i;

#define GAT_B    2
#define GAT_N    2048
#define GAT_INF  512
#define GAT_OUTF 64
#define GAT_H    4
#define GAT_COLS 256      // GAT_H * GAT_OUTF
#define GAT_ALPHA 0.2f
#define NEG_BIG  (-3.0e38f)

// ---------------------------------------------------------------------------
// Kernel 1: h = x @ W  (f16 WMMA, f32 accumulate)
// ---------------------------------------------------------------------------
__global__ __launch_bounds__(256) void gemm_h_kernel(
    const float* __restrict__ x, const float* __restrict__ W,
    float* __restrict__ h) {
  const int wid   = blockIdx.x * 8 + (threadIdx.x >> 5);  // 0..4095
  const int lane  = threadIdx.x & 31;
  const int mtile = wid >> 4;
  const int ntile = wid & 15;
  const int loc   = lane & 15;
  const int half  = lane >> 4;
  const int c0    = half * 8;

  const int row  = mtile * 16 + loc;
  const int colW = ntile * 16 + loc;
  const float* xrow = x + (size_t)row * GAT_INF;

  v8f acc = {};
  for (int kb = 0; kb < GAT_INF; kb += 32) {
    const float4 x0 = *(const float4*)(xrow + kb + c0);
    const float4 x1 = *(const float4*)(xrow + kb + c0 + 4);
    const float4 x2 = *(const float4*)(xrow + kb + 16 + c0);
    const float4 x3 = *(const float4*)(xrow + kb + 16 + c0 + 4);
    v16h a;
    a[0]=(_Float16)x0.x; a[1]=(_Float16)x0.y; a[2]=(_Float16)x0.z; a[3]=(_Float16)x0.w;
    a[4]=(_Float16)x1.x; a[5]=(_Float16)x1.y; a[6]=(_Float16)x1.z; a[7]=(_Float16)x1.w;
    a[8]=(_Float16)x2.x; a[9]=(_Float16)x2.y; a[10]=(_Float16)x2.z; a[11]=(_Float16)x2.w;
    a[12]=(_Float16)x3.x; a[13]=(_Float16)x3.y; a[14]=(_Float16)x3.z; a[15]=(_Float16)x3.w;
    v16h b;
#pragma unroll
    for (int e = 0; e < 16; ++e)
      b[e] = (_Float16)W[(size_t)(kb + c0 + e + (e >= 8 ? 8 : 0)) * GAT_COLS + colW];
    acc = __builtin_amdgcn_wmma_f32_16x16x32_f16(false, a, false, b,
                                                 (short)0, acc, false, false);
  }
#pragma unroll
  for (int r = 0; r < 8; ++r) {
    const int mrow = mtile * 16 + half * 8 + r;
    h[(size_t)mrow * GAT_COLS + ntile * 16 + loc] = acc[r];
  }
}

// ---------------------------------------------------------------------------
// Kernel 2: e_src/e_dst dots + f16 transpose of h into hT[b,h,f,i]
// ---------------------------------------------------------------------------
__global__ __launch_bounds__(256) void prep_kernel(
    const float* __restrict__ h, const float* __restrict__ avec,
    float* __restrict__ e_src, float* __restrict__ e_dst,
    _Float16* __restrict__ hT) {
  const int bh   = blockIdx.x;
  const int b    = bh >> 2;
  const int head = bh & 3;
  const int n    = blockIdx.y * 256 + threadIdx.x;
  const float* hp = h + ((size_t)(b * GAT_N + n)) * GAT_COLS + head * GAT_OUTF;
  _Float16* tp = hT + ((size_t)bh * GAT_OUTF) * GAT_N + n;
  float es = 0.f, ed = 0.f;
#pragma unroll
  for (int f = 0; f < GAT_OUTF; f += 4) {
    const float4 hv = *(const float4*)(hp + f);
    es += hv.x*avec[f] + hv.y*avec[f+1] + hv.z*avec[f+2] + hv.w*avec[f+3];
    ed += hv.x*avec[GAT_OUTF+f]   + hv.y*avec[GAT_OUTF+f+1]
        + hv.z*avec[GAT_OUTF+f+2] + hv.w*avec[GAT_OUTF+f+3];
    tp[(size_t)(f+0) * GAT_N] = (_Float16)hv.x;
    tp[(size_t)(f+1) * GAT_N] = (_Float16)hv.y;
    tp[(size_t)(f+2) * GAT_N] = (_Float16)hv.z;
    tp[(size_t)(f+3) * GAT_N] = (_Float16)hv.w;
  }
  e_src[(b * GAT_N + n) * GAT_H + head] = es;
  e_dst[(b * GAT_N + n) * GAT_H + head] = ed;
}

// ---------------------------------------------------------------------------
// Kernel 3: online softmax statistics along i for each (b,head,j) column.
// ---------------------------------------------------------------------------
__global__ __launch_bounds__(256) void stats_kernel(
    const int* __restrict__ adj, const float* __restrict__ e_src,
    const float* __restrict__ e_dst, float* __restrict__ mbuf,
    float* __restrict__ lbuf) {
  const int idx  = blockIdx.x * 256 + threadIdx.x;
  const int j    = idx & (GAT_N - 1);
  const int bh   = idx >> 11;
  const int b    = bh >> 2;
  const int head = bh & 3;
  const float ed = e_dst[(b * GAT_N + j) * GAT_H + head];
  const int*   ac  = adj + (size_t)b * GAT_N * GAT_N + j;
  const float* esp = e_src + (size_t)b * GAT_N * GAT_H + head;
  float m = NEG_BIG, l = 0.f;
  for (int i = 0; i < GAT_N; ++i) {
    if (ac[(size_t)i * GAT_N] > 0) {
      float s = esp[i * GAT_H] + ed;
      s = (s >= 0.f) ? s : GAT_ALPHA * s;
      const float mn = fmaxf(m, s);
      l = l * __expf(m - mn) + __expf(s - mn);
      m = mn;
    }
  }
  mbuf[idx] = m;
  lbuf[idx] = l;
}

// ---------------------------------------------------------------------------
// TDM: load one 64(f) x 32(i) f16 chunk of hT into LDS, padding each 64B row
// with 16B so LDS row stride is 80B (bank-conflict-free ds_load_b128).
// D# group0: count=1, lds_addr, global_addr, type=2.
// D# group1: data_size=2B, pad_enable, pad_interval=16dw, pad_amount=4dw,
//            tensor_dim0=2048, tensor_dim1=64, tile_dim0=32, tile_dim1=64,
//            tensor_dim0_stride=2048.
// ---------------------------------------------------------------------------
__device__ inline void tdm_load_chunk(const _Float16* gsrc, unsigned int lds_off) {
  const unsigned long long ga = (unsigned long long)(uintptr_t)gsrc;
  v4u g0;
  g0[0] = 1u;                                   // count=1, no gather
  g0[1] = lds_off;                              // LDS byte address
  g0[2] = (unsigned int)ga;                     // global addr [31:0]
  g0[3] = ((unsigned int)(ga >> 32) & 0x01FFFFFFu) | (2u << 30); // addr[56:32], type=2
  v8i g1;
  g1[0] = (1 << 16) | (1 << 20) | (3 << 22) | (3 << 25); // 2B elems, pad 64B->+16B
  g1[1] = (int)(2048u << 16);                   // tensor_dim0 = 2048 (low 16)
  g1[2] = (int)(64u << 16);                     // dim0 hi=0, tensor_dim1 = 64
  g1[3] = (int)(32u << 16);                     // dim1 hi=0, tile_dim0 = 32
  g1[4] = 64;                                   // tile_dim1 = 64, tile_dim2 = 0
  g1[5] = 2048;                                 // tensor_dim0_stride lo
  g1[6] = 0;                                    // stride hi, dim1_stride lo
  g1[7] = 0;
  v4i z = {0, 0, 0, 0};
#if defined(__clang_major__) && (__clang_major__ >= 23)
  v8i z8 = {0, 0, 0, 0, 0, 0, 0, 0};
  __builtin_amdgcn_tensor_load_to_lds(g0, g1, z, z, z8, 0);
#else
  __builtin_amdgcn_tensor_load_to_lds(g0, g1, z, z, 0);
#endif
}

// ---------------------------------------------------------------------------
// Kernel 4: out[j,f] = (1/l_j) * sum_i exp(score(i,j)-m_j) * h[i,f]
// Block = 8 waves sharing one (b,head); hT chunks staged in LDS by the TDM
// (double-buffered), A = P^T built on the fly, 4 WMMAs per chunk per wave.
// ---------------------------------------------------------------------------
__global__ __launch_bounds__(256) void attn_kernel(
    const int* __restrict__ adj, const float* __restrict__ e_src,
    const float* __restrict__ e_dst, const float* __restrict__ mbuf,
    const float* __restrict__ lbuf, const _Float16* __restrict__ hT,
    float* __restrict__ out) {
  __shared__ __align__(32) _Float16 sbuf[2][64 * 40];  // 80B row stride, 5120B/buf

  const int w      = threadIdx.x >> 5;
  const int lane   = threadIdx.x & 31;
  const int blk    = blockIdx.x;        // 0..127
  const int jblock = blk & 15;
  const int head   = (blk >> 4) & 3;
  const int b      = blk >> 6;
  const int bh     = b * GAT_H + head;
  const int jtile  = jblock * 8 + w;
  const int loc    = lane & 15;
  const int half   = lane >> 4;
  const int c0     = half * 8;
  const int jg     = jtile * 16 + loc;  // A-matrix row (j) for this lane

  const float ed = e_dst[(b * GAT_N + jg) * GAT_H + head];
  const float mj = mbuf[bh * GAT_N + jg];
  const int*      adjb = adj + (size_t)b * GAT_N * GAT_N;
  const float*    esp  = e_src + (size_t)b * GAT_N * GAT_H + head;
  const _Float16* hTb  = hT + (size_t)bh * GAT_OUTF * GAT_N;

  // LDS aperture maps flat addr[31:0] -> group-segment byte offset (ISA 10.2)
  const unsigned int lds0 = (unsigned int)(uintptr_t)&sbuf[0][0];
  const unsigned int lds1 = (unsigned int)(uintptr_t)&sbuf[1][0];

  v8f acc[4] = {{}, {}, {}, {}};

  if (w == 0) tdm_load_chunk(hTb, lds0);          // preload chunk 0

  for (int ic = 0; ic < 64; ++ic) {
    const int ib = ic * 32;
    if (w == 0) {
      if (ic + 1 < 64) {
        tdm_load_chunk(hTb + ib + 32, ((ic + 1) & 1) ? lds1 : lds0);
        __builtin_amdgcn_s_wait_tensorcnt(1);     // chunk ic landed (in-order)
      } else {
        __builtin_amdgcn_s_wait_tensorcnt(0);
      }
    }
    __syncthreads();                              // publish chunk ic to all waves

    // A = P^T tile: element e<8 -> i = ib+c0+e ; e>=8 -> i = ib+16+c0+(e-8)
    v16h pa;
#pragma unroll
    for (int e = 0; e < 8; ++e) {
      const int i0 = ib + c0 + e;
      const int i1 = ib + 16 + c0 + e;
      float s0 = esp[i0 * GAT_H] + ed;  s0 = (s0 >= 0.f) ? s0 : GAT_ALPHA * s0;
      float s1 = esp[i1 * GAT_H] + ed;  s1 = (s1 >= 0.f) ? s1 : GAT_ALPHA * s1;
      const float p0 = (adjb[(size_t)i0 * GAT_N + jg] > 0) ? __expf(s0 - mj) : 0.f;
      const float p1 = (adjb[(size_t)i1 * GAT_N + jg] > 0) ? __expf(s1 - mj) : 0.f;
      pa[e]     = (_Float16)p0;
      pa[e + 8] = (_Float16)p1;
    }

    const _Float16* base = &sbuf[ic & 1][0];
#pragma unroll
    for (int ft = 0; ft < 4; ++ft) {
      // B tile: lane col f = ft*16+loc, element e<8 -> i = ib+c0+e ; e>=8 -> i = ib+16+c0+(e-8)
      const _Float16* sp = base + (ft * 16 + loc) * 40 + half * 8;
      const v8h b0 = *(const v8h*)sp;
      const v8h b1 = *(const v8h*)(sp + 16);
      const v16h pb = __builtin_shufflevector(
          b0, b1, 0, 1, 2, 3, 4, 5, 6, 7, 8, 9, 10, 11, 12, 13, 14, 15);
      acc[ft] = __builtin_amdgcn_wmma_f32_16x16x32_f16(false, pa, false, pb,
                                                       (short)0, acc[ft],
                                                       false, false);
    }
    __syncthreads();                 // buffer reads done before TDM overwrites
  }

  float* ob = out + (size_t)b * GAT_N * GAT_COLS + head * GAT_OUTF;
#pragma unroll
  for (int r = 0; r < 8; ++r) {
    const int jr = jtile * 16 + half * 8 + r;     // D-matrix row for VGPR r
    const float lv = lbuf[bh * GAT_N + jr];
    const float sc = (lv > 0.f) ? (1.f / lv) : 0.f;
#pragma unroll
    for (int ft = 0; ft < 4; ++ft)
      ob[(size_t)jr * GAT_COLS + ft * 16 + loc] = acc[ft][r] * sc;
  }
}

// ---------------------------------------------------------------------------
extern "C" void kernel_launch(void* const* d_in, const int* in_sizes, int n_in,
                              void* d_out, int out_size, void* d_ws, size_t ws_size,
                              hipStream_t stream) {
  const float* x   = (const float*)d_in[0];  // (B,N,IN_F)
  const int*   adj = (const int*)d_in[1];    // (B,N,N)
  const float* W   = (const float*)d_in[2];  // (IN_F, OUT_F*H)
  const float* a   = (const float*)d_in[3];  // (2*OUT_F, 1)
  float* out = (float*)d_out;                // (B,N,H*OUT_F)

  float* h     = (float*)d_ws;                          // 4096*256 f32
  float* e_src = h + (size_t)GAT_B * GAT_N * GAT_COLS;  // 16384
  float* e_dst = e_src + GAT_B * GAT_N * GAT_H;
  float* mbuf  = e_dst + GAT_B * GAT_N * GAT_H;
  float* lbuf  = mbuf + GAT_B * GAT_H * GAT_N;
  _Float16* hT = (_Float16*)(lbuf + GAT_B * GAT_H * GAT_N); // 2*4*64*2048 f16

  gemm_h_kernel<<<512, 256, 0, stream>>>(x, W, h);
  prep_kernel<<<dim3(GAT_B * GAT_H, GAT_N / 256), 256, 0, stream>>>(
      h, a, e_src, e_dst, hT);
  stats_kernel<<<GAT_B * GAT_H * GAT_N / 256, 256, 0, stream>>>(
      adj, e_src, e_dst, mbuf, lbuf);
  attn_kernel<<<GAT_B * GAT_H * (GAT_N / 16) / 16 * 2, 256, 0, stream>>>(
      adj, e_src, e_dst, mbuf, lbuf, hT, out);
}
